// Batch_GraphConv_3676492005534
// MI455X (gfx1250) — matrix operand order repaired and checked by reading, not compiled, not run on hardware
//
#include <hip/hip_runtime.h>
#include <hip/hip_bf16.h>

// B=64, N=512, D=64, S=2, MAX_STEP=2 -> metrics m0..m4, OUT=64
#define GN 512
#define GD 64
#define GB 64

typedef __attribute__((ext_vector_type(16))) __bf16 v16bf;
typedef __attribute__((ext_vector_type(8)))  float  v8f;
typedef int v4i_vs __attribute__((vector_size(16)));
#define AS1 __attribute__((address_space(1)))
#define AS3 __attribute__((address_space(3)))

union BF16x16 {
    v16bf v;
    unsigned int u[8];
};

// Pack two f32 -> two bf16 in one VALU op where possible.
__device__ __forceinline__ unsigned int pk2(float lo, float hi) {
#if __has_builtin(__builtin_amdgcn_cvt_pk_bf16_f32)
    typedef __attribute__((ext_vector_type(2))) __bf16 v2bf;
    union { v2bf v; unsigned int u; } c;
    c.v = __builtin_amdgcn_cvt_pk_bf16_f32(lo, hi);   // RNE, single v_cvt_pk
    return c.u;
#else
    // single v_perm_b32: truncate-to-bf16 {hi[31:16], lo[31:16]}
    return __builtin_amdgcn_perm(__float_as_uint(hi), __float_as_uint(lo),
                                 0x07060302u);
#endif
}

// -------------------------------------------------------------------------
// Stage GEMM: per batch b, Y = A @ X   (USEPREV: Y = 2*(A@X) - Xprev)
// A: [512,512] fp32 row-major, X/Y: [512,64] fp32 row-major.
// grid = (N/128, B), block = 256 (8 waves, 16 output rows per wave).
// X staged transposed+bf16 in LDS: Xt[d][k], padded stride 520 (bank-clean
// for the per-lane-column ds_load_b128 B-fragment reads).
// -------------------------------------------------------------------------
template <int USEPREV>
__global__ __launch_bounds__(256) void stage_kernel(
    const float* __restrict__ Abase, const float* __restrict__ X,
    const float* __restrict__ Xprev, float* __restrict__ Y)
{
    __shared__ unsigned short Xt[GD * 520];   // 66,560 bytes

    const int b = blockIdx.y;
    const float* A   = Abase + (size_t)b * GN * GN;
    const float* Xb  = X     + (size_t)b * GN * GD;
    const float* Xpb = Xprev + (size_t)b * GN * GD;
    float*       Yb  = Y     + (size_t)b * GN * GD;

    // ---- preload: X (fp32 row-major) -> Xt (bf16, transposed) ----
    {
        const int t  = threadIdx.x;
        const int d2 = (t & 31) * 2;          // two adjacent columns of X
        const int kb = (t >> 5) * 64;         // 8 wave-groups cover k
        for (int kk = 0; kk < 64; kk += 2) {
            float2 r0 = *(const float2*)(Xb + (size_t)(kb + kk) * GD + d2);
            float2 r1 = *(const float2*)(Xb + (size_t)(kb + kk + 1) * GD + d2);
            *(unsigned int*)(&Xt[(d2 + 0) * 520 + kb + kk]) = pk2(r0.x, r1.x);
            *(unsigned int*)(&Xt[(d2 + 1) * 520 + kb + kk]) = pk2(r0.y, r1.y);
        }
    }
    __syncthreads();

    const int lane = threadIdx.x & 31;
    const int wave = threadIdx.x >> 5;
    const int m0   = blockIdx.x * 128 + wave * 16;
    const int nloc = lane & 15;
    const int h    = lane >> 4;              // lane-half selects K sub-chunk
    const float* arow = A + (size_t)(m0 + nloc) * GN;

    v8f acc[4];
    #pragma unroll
    for (int ct = 0; ct < 4; ++ct) acc[ct] = (v8f){0.f,0.f,0.f,0.f,0.f,0.f,0.f,0.f};

    for (int k = 0; k < GN; k += 32) {
        // A fragment 16x32 bf16: lane<16 -> K {k..k+7, k+16..k+23}; lane>=16 -> +8
        const float* ap = arow + k + h * 8;
        __builtin_prefetch(ap + 128, 0, 3);   // stream A ahead (global_prefetch)
        float4 a0 = *(const float4*)(ap);
        float4 a1 = *(const float4*)(ap + 4);
        float4 a2 = *(const float4*)(ap + 16);
        float4 a3 = *(const float4*)(ap + 20);
        BF16x16 af;
        af.u[0] = pk2(a0.x, a0.y); af.u[1] = pk2(a0.z, a0.w);
        af.u[2] = pk2(a1.x, a1.y); af.u[3] = pk2(a1.z, a1.w);
        af.u[4] = pk2(a2.x, a2.y); af.u[5] = pk2(a2.z, a2.w);
        af.u[6] = pk2(a3.x, a3.y); af.u[7] = pk2(a3.z, a3.w);

        #pragma unroll
        for (int ct = 0; ct < 4; ++ct) {
            // B fragment 32x16 bf16: lane -> n = lane&15, elements -> K {8h..8h+7, 16+8h..16+8h+7}
            const unsigned short* tp = &Xt[(ct * 16 + nloc) * 520 + k + 8 * h];
            uint4 b0 = *(const uint4*)(tp);
            uint4 b1 = *(const uint4*)(tp + 16);
            BF16x16 bf;
            bf.u[0] = b0.x; bf.u[1] = b0.y; bf.u[2] = b0.z; bf.u[3] = b0.w;
            bf.u[4] = b1.x; bf.u[5] = b1.y; bf.u[6] = b1.z; bf.u[7] = b1.w;
            acc[ct] = __builtin_amdgcn_wmma_f32_16x16x32_bf16(
                false, af.v, false, bf.v, (short)0, acc[ct], false, false);
        }
    }

    // epilogue: C/D layout -> row = m0 + r + 8*h, col = ct*16 + nloc
    float*       yp = Yb  + (size_t)(m0 + 8 * h) * GD + nloc;
    const float* pp = Xpb + (size_t)(m0 + 8 * h) * GD + nloc;
    #pragma unroll
    for (int ct = 0; ct < 4; ++ct) {
        #pragma unroll
        for (int r = 0; r < 8; ++r) {
            float v = acc[ct][r];
            if (USEPREV) v = fmaf(v, 2.0f, -pp[r * GD + ct * 16]);
            yp[r * GD + ct * 16] = v;
        }
    }
}

// -------------------------------------------------------------------------
// W repack: Wt[(m*64 + n)*64 + d] = bf16(W[(d*5 + m)*64 + n])
// (stored as uint pairs over d)
// -------------------------------------------------------------------------
__global__ __launch_bounds__(256) void convw_kernel(
    const float* __restrict__ W, unsigned int* __restrict__ Wt)
{
    int i = blockIdx.x * 256 + threadIdx.x;     // i over pairs (2 bf16 per uint)
    if (i >= 5 * 64 * 32) return;
    int d = (i & 31) * 2;
    int n = (i >> 5) & 63;
    int m = i >> 11;
    float lo = W[((size_t)d * 5 + m) * 64 + n];
    float hi = W[((size_t)(d + 1) * 5 + m) * 64 + n];
    Wt[i] = pk2(lo, hi);
}

// -------------------------------------------------------------------------
// Projection: out[r, :] = sum_m metric_m[r, :] @ Wm + bias ; r over B*N=32768
// grid = 256, block = 256 (8 waves; each wave one 16-row tile x 64 cols).
// Wt (40 KB bf16) is staged once per workgroup into LDS (padded stride 72)
// via async global->LDS loads when available.
// -------------------------------------------------------------------------
__global__ __launch_bounds__(256) void proj_kernel(
    const float* __restrict__ M0, const float* __restrict__ M1,
    const float* __restrict__ M2, const float* __restrict__ M3,
    const float* __restrict__ M4, const unsigned short* __restrict__ Wt,
    const float* __restrict__ bias, float* __restrict__ out)
{
    __shared__ unsigned short Ws[5 * 64 * 72];   // 46,080 bytes, 320 rows x 72

    // ---- stage Wt into LDS: 320 rows of 64 bf16 = 2560 x 16B chunks ----
#if __has_builtin(__builtin_amdgcn_global_load_async_to_lds_b128)
    for (int c = threadIdx.x; c < 2560; c += 256) {
        const int row = c >> 3, q = c & 7;
        __builtin_amdgcn_global_load_async_to_lds_b128(
            (AS1 v4i_vs*)(Wt + row * 64 + q * 8),
            (AS3 v4i_vs*)&Ws[row * 72 + q * 8],
            0, 0);
    }
#if __has_builtin(__builtin_amdgcn_s_wait_asynccnt)
    __builtin_amdgcn_s_wait_asynccnt(0);
#else
    asm volatile("s_wait_asynccnt 0" ::: "memory");
#endif
#else
    for (int c = threadIdx.x; c < 2560; c += 256) {
        const int row = c >> 3, q = c & 7;
        *(uint4*)&Ws[row * 72 + q * 8] = *(const uint4*)(Wt + row * 64 + q * 8);
    }
#endif
    __syncthreads();

    const float* Ms[5] = {M0, M1, M2, M3, M4};
    const int lane = threadIdx.x & 31;
    const int wave = threadIdx.x >> 5;
    const int row0 = (blockIdx.x * 8 + wave) * 16;
    const int nloc = lane & 15;
    const int h    = lane >> 4;
    const int r    = row0 + nloc;

    v8f acc[4];
    #pragma unroll
    for (int ct = 0; ct < 4; ++ct) acc[ct] = (v8f){0.f,0.f,0.f,0.f,0.f,0.f,0.f,0.f};

    #pragma unroll
    for (int m = 0; m < 5; ++m) {
        const float* src = Ms[m] + (size_t)r * GD;
        #pragma unroll
        for (int ks = 0; ks < 64; ks += 32) {
            const float* ap = src + ks + h * 8;
            float4 a0 = *(const float4*)(ap);
            float4 a1 = *(const float4*)(ap + 4);
            float4 a2 = *(const float4*)(ap + 16);
            float4 a3 = *(const float4*)(ap + 20);
            BF16x16 af;
            af.u[0] = pk2(a0.x, a0.y); af.u[1] = pk2(a0.z, a0.w);
            af.u[2] = pk2(a1.x, a1.y); af.u[3] = pk2(a1.z, a1.w);
            af.u[4] = pk2(a2.x, a2.y); af.u[5] = pk2(a2.z, a2.w);
            af.u[6] = pk2(a3.x, a3.y); af.u[7] = pk2(a3.z, a3.w);

            #pragma unroll
            for (int ct = 0; ct < 4; ++ct) {
                const unsigned short* tp =
                    &Ws[(m * 64 + ct * 16 + nloc) * 72 + ks + 8 * h];
                uint4 b0 = *(const uint4*)(tp);
                uint4 b1 = *(const uint4*)(tp + 16);
                BF16x16 bf;
                bf.u[0] = b0.x; bf.u[1] = b0.y; bf.u[2] = b0.z; bf.u[3] = b0.w;
                bf.u[4] = b1.x; bf.u[5] = b1.y; bf.u[6] = b1.z; bf.u[7] = b1.w;
                acc[ct] = __builtin_amdgcn_wmma_f32_16x16x32_bf16(
                    false, af.v, false, bf.v, (short)0, acc[ct], false, false);
            }
        }
    }

    float* op = out + (size_t)(row0 + 8 * h) * GD + nloc;
    #pragma unroll
    for (int ct = 0; ct < 4; ++ct) {
        const float bn = bias[ct * 16 + nloc];
        #pragma unroll
        for (int rr = 0; rr < 8; ++rr) {
            op[rr * GD + ct * 16] = acc[ct][rr] + bn;
        }
    }
}

extern "C" void kernel_launch(void* const* d_in, const int* in_sizes, int n_in,
                              void* d_out, int out_size, void* d_ws, size_t ws_size,
                              hipStream_t stream) {
    const float* inputs   = (const float*)d_in[0];   // [64,512,64]
    const float* supports = (const float*)d_in[1];   // [2,64,512,512]
    const float* W        = (const float*)d_in[2];   // [320,64]
    const float* bias     = (const float*)d_in[3];   // [64]
    float* out = (float*)d_out;

    const size_t MSZ = (size_t)GB * GN * GD;         // 2,097,152 floats = 8 MB
    float* m1 = (float*)d_ws;
    float* m2 = m1 + MSZ;
    float* m3 = m2 + MSZ;
    float* m4 = m3 + MSZ;
    unsigned int* Wt = (unsigned int*)(m4 + MSZ);    // +40 KB (bf16 pairs)

    convw_kernel<<<(5 * 64 * 32 + 255) / 256, 256, 0, stream>>>(W, Wt);

    const float* A0 = supports;
    const float* A1 = supports + (size_t)GB * GN * GN;
    dim3 sg(GN / 128, GB);
    // m1 = A0 @ m0
    stage_kernel<0><<<sg, 256, 0, stream>>>(A0, inputs, inputs, m1);
    // m2 = 2*A0 @ m1 - m0
    stage_kernel<1><<<sg, 256, 0, stream>>>(A0, m1, inputs, m2);
    // m3 = A1 @ m1
    stage_kernel<0><<<sg, 256, 0, stream>>>(A1, m1, inputs, m3);
    // m4 = 2*A1 @ m3 - m1
    stage_kernel<1><<<sg, 256, 0, stream>>>(A1, m3, m1, m4);

    proj_kernel<<<256, 256, 0, stream>>>(inputs, m1, m2, m3, m4,
                                         (const unsigned short*)Wt, bias, out);
}
